// down_12524124635534
// MI455X (gfx1250) — matrix operand order repaired and checked by reading, not compiled, not run on hardware
//
#include <hip/hip_runtime.h>
#include <hip/hip_bf16.h>

// ---------------- problem constants (match reference) ----------------
#define NN      20000
#define EE      640000
#define DD      3
#define PP      27          // 3^3 kernel bins
#define CIN     64
#define COUT    64
#define MM      5000        // FPS samples
#define RRAD    0.1f
#define MAXN    64
#define KTOT    (PP * CIN)  // 1728 = GEMM reduction dim
#define KSTEPS  (KTOT / 32) // 54 bf16-WMMA k-steps
#define ACC_ELEMS ((size_t)NN * PP * CIN)

// ---------------- WMMA vector types ----------------
typedef __attribute__((ext_vector_type(16))) __bf16 v16bf;
typedef __attribute__((ext_vector_type(8)))  float  v8f;
typedef __attribute__((ext_vector_type(8)))  unsigned int v8u;
// GCC-style vector for the async-LDS builtin signature (diag showed
// '__attribute__((__vector_size__(4 * sizeof(int)))) int *').
typedef int v4i_g __attribute__((vector_size(16)));

__device__ __forceinline__ unsigned short f2bfbits(float f) {
    unsigned u = __builtin_bit_cast(unsigned, f);
    unsigned r = u + 0x7FFFu + ((u >> 16) & 1u);   // RNE to bf16
    return (unsigned short)(r >> 16);
}
__device__ __forceinline__ __bf16 bits2bf(unsigned short s) {
    return __builtin_bit_cast(__bf16, s);
}

// ---------------- gfx1250 async global->LDS staging ----------------
#define GLOBAL_AS __attribute__((address_space(1)))
#define LDS_AS    __attribute__((address_space(3)))

#if __has_builtin(__builtin_amdgcn_global_load_async_to_lds_b128)
#define HAVE_ASYNC_LDS 1
#else
#define HAVE_ASYNC_LDS 0
#endif

// one 16-byte lane transfer: global -> LDS
__device__ __forceinline__ void stage_b128(const float* gsrc, float* ldst) {
#if HAVE_ASYNC_LDS
    __builtin_amdgcn_global_load_async_to_lds_b128(
        (GLOBAL_AS v4i_g*)(uintptr_t)gsrc,
        (LDS_AS    v4i_g*)(uintptr_t)ldst, 0, 0);
#else
    *(float4*)ldst = *(const float4*)gsrc;
#endif
}

__device__ __forceinline__ void async_wait(int pending) {
#if HAVE_ASYNC_LDS
#if __has_builtin(__builtin_amdgcn_s_wait_asynccnt)
    if (pending == 0) __builtin_amdgcn_s_wait_asynccnt(0);
    else              __builtin_amdgcn_s_wait_asynccnt(1);
#else
    if (pending == 0) asm volatile("s_wait_asynccnt 0" ::: "memory");
    else              asm volatile("s_wait_asynccnt 1" ::: "memory");
#endif
#else
    (void)pending;
#endif
}

// ---------------- WMMA fragment builders ----------------
// A 16x32 bf16 (wave32): lane row = lane&15; lo lanes hold k 0..15,
// hi lanes hold k 16..31 (matches B) -> one contiguous 16-float run at
// off0 = hi?16:0.  sA row stride = 36 floats.
__device__ __forceinline__ v16bf load_a_frag(const float* sA, int row, int hi) {
    const float* base = sA + row * 36 + (hi ? 16 : 0);
    float4 f0 = *(const float4*)(base + 0);
    float4 f1 = *(const float4*)(base + 4);
    float4 f2 = *(const float4*)(base + 8);
    float4 f3 = *(const float4*)(base + 12);
    float t[16] = {f0.x, f0.y, f0.z, f0.w, f1.x, f1.y, f1.z, f1.w,
                   f2.x, f2.y, f2.z, f2.w, f3.x, f3.y, f3.z, f3.w};
    v16bf a;
#pragma unroll
    for (int j = 0; j < 16; ++j) a[j] = bits2bf(f2bfbits(t[j]));
    return a;
}

// B 32x16 bf16 (wave32): lane col = lane&15; lo lanes k=0..15, hi lanes
// k=16..31, consecutive per lane -> 16 consecutive bf16 from transposed
// pre-packed weights = two uint4 loads.
__device__ __forceinline__ v16bf load_b_frag(const unsigned short* p) {
    uint4 b0 = *(const uint4*)p;
    uint4 b1 = *(const uint4*)(p + 8);
    v8u r = {b0.x, b0.y, b0.z, b0.w, b1.x, b1.y, b1.z, b1.w};
    return __builtin_bit_cast(v16bf, r);
}

// ====================================================================
// Kernel 0: zero the accumulator + degree arrays (vectorized)
// ====================================================================
__global__ void zero_ws_kernel(float4* p, size_t n4) {
    size_t i = (size_t)blockIdx.x * blockDim.x + threadIdx.x;
    size_t stride = (size_t)gridDim.x * blockDim.x;
    float4 z = {0.0f, 0.0f, 0.0f, 0.0f};
    for (; i < n4; i += stride) p[i] = z;
}

// ====================================================================
// Kernel 0b: pre-pack weights to transposed bf16 (done once per launch;
// 221 KB -> L2 resident for all 1250 GEMM tiles).
//   wT[col*1728 + k]  = bf16(w[k*64 + col])
//   rwT[col*64 + k]   = bf16(rw[k*64 + col])
// ====================================================================
__global__ void pack_weights_kernel(const float* __restrict__ w,
                                    const float* __restrict__ rw,
                                    unsigned short* __restrict__ wT,
                                    unsigned short* __restrict__ rwT) {
    int t = blockIdx.x * blockDim.x + threadIdx.x;
    if (t < KTOT * COUT) {
        int col = t / KTOT, k = t % KTOT;
        wT[t] = f2bfbits(w[(size_t)k * COUT + col]);
    }
    int t2 = t - KTOT * COUT;
    if (t2 >= 0 && t2 < CIN * COUT) {
        int col = t2 / CIN, k = t2 % CIN;
        rwT[t2] = f2bfbits(rw[(size_t)k * COUT + col]);
    }
}

// ====================================================================
// Kernel 1: spline-basis edge scatter. One wave32 per edge; each lane
// owns 2 channels -> coalesced global_atomic_add_f32.
// ====================================================================
__global__ void spline_scatter_kernel(const float* __restrict__ x,
                                      const int*   __restrict__ ei,
                                      const float* __restrict__ ea,
                                      float* __restrict__ acc,
                                      float* __restrict__ deg) {
    long long gid = (long long)blockIdx.x * blockDim.x + threadIdx.x;
    int e    = (int)(gid >> 5);
    int lane = (int)(gid & 31);
    if (e >= EE) return;

    int src = ei[e];
    int dst = ei[EE + e];

    float f[DD]; int kb[DD];
#pragma unroll
    for (int d = 0; d < DD; ++d) {
        float v  = ea[e * DD + d] * 2.0f;          // pseudo * (K-1)
        float fl = floorf(v);
        fl = fminf(fmaxf(fl, 0.0f), 1.0f);         // clip to [0, K-2]
        f[d]  = v - fl;
        kb[d] = (int)fl;
    }

    float x0 = x[(size_t)src * CIN + lane];
    float x1 = x[(size_t)src * CIN + lane + 32];

    if (lane == 0) atomicAdd(&deg[dst], 1.0f);

    float* accd = acc + (size_t)dst * KTOT;
    const int strides[3] = {9, 3, 1};
#pragma unroll
    for (int s = 0; s < 8; ++s) {
        float b = 1.0f; int pidx = 0;
#pragma unroll
        for (int d = 0; d < DD; ++d) {
            int bit = (s >> d) & 1;
            b *= bit ? f[d] : (1.0f - f[d]);
            pidx += (kb[d] + bit) * strides[d];
        }
        float* row = accd + pidx * CIN;
        atomicAdd(&row[lane],      x0 * b);
        atomicAdd(&row[lane + 32], x1 * b);
    }
}

// ====================================================================
// Kernel 2: farthest point sampling. Single 1024-thread workgroup,
// register-resident points + min-dist, wave32 shuffle argmax + LDS tree.
// ====================================================================
#define FPS_PTS 20
__global__ void __launch_bounds__(1024)
fps_kernel(const float* __restrict__ pos, int* __restrict__ idx) {
    __shared__ float s_val[32];
    __shared__ int   s_idx[32];
    __shared__ int   s_next;

    int tid  = threadIdx.x;
    int wid  = tid >> 5;
    int lane = tid & 31;

    float px[FPS_PTS], py[FPS_PTS], pz[FPS_PTS], mind[FPS_PTS];
    float q0x = pos[0], q0y = pos[1], q0z = pos[2];
#pragma unroll
    for (int t = 0; t < FPS_PTS; ++t) {
        int i = tid + t * 1024;
        if (i < NN) {
            px[t] = pos[i * 3 + 0];
            py[t] = pos[i * 3 + 1];
            pz[t] = pos[i * 3 + 2];
            float dx = px[t] - q0x, dy = py[t] - q0y, dz = pz[t] - q0z;
            mind[t] = dx * dx + dy * dy + dz * dz;
        } else {
            px[t] = py[t] = pz[t] = 0.0f;
            mind[t] = -1e30f;                      // never selected
        }
    }
    if (tid == 0) idx[0] = 0;

    for (int it = 1; it < MM; ++it) {
        float bm = -1e38f; int bi = 0x7fffffff;
#pragma unroll
        for (int t = 0; t < FPS_PTS; ++t) {
            int i = tid + t * 1024;
            float v = mind[t];
            if (v > bm || (v == bm && i < bi)) { bm = v; bi = i; }
        }
#pragma unroll
        for (int off = 16; off > 0; off >>= 1) {
            float ov = __shfl_down(bm, (unsigned)off);
            int   oi = __shfl_down(bi, (unsigned)off);
            if (ov > bm || (ov == bm && oi < bi)) { bm = ov; bi = oi; }
        }
        if (lane == 0) { s_val[wid] = bm; s_idx[wid] = bi; }
        __syncthreads();
        if (wid == 0) {
            float v = s_val[lane]; int ii = s_idx[lane];
#pragma unroll
            for (int off = 16; off > 0; off >>= 1) {
                float ov = __shfl_down(v, (unsigned)off);
                int   oi = __shfl_down(ii, (unsigned)off);
                if (ov > v || (ov == v && oi < ii)) { v = ov; ii = oi; }
            }
            if (lane == 0) { s_next = ii; idx[it] = ii; }
        }
        __syncthreads();
        int nxt = s_next;
        float qx = pos[nxt * 3 + 0], qy = pos[nxt * 3 + 1], qz = pos[nxt * 3 + 2];
#pragma unroll
        for (int t = 0; t < FPS_PTS; ++t) {
            float dx = px[t] - qx, dy = py[t] - qy, dz = pz[t] - qz;
            float d2 = dx * dx + dy * dy + dz * dz;
            mind[t] = fminf(mind[t], d2);
        }
    }
}

// ====================================================================
// Kernel 3: SplineConv GEMM via bf16 WMMA.
//   out[n,o] = (acc[n,:] . W[:,o]) / deg[n] + (x @ root_w)[n,o] + bias[o]
//   then ELU.
// Block = 128 threads (4 waves) owns a 16-node tile; wave w owns output
// columns [16w,16w+16).  A-tile (16x32 f32, 2 KB) double-buffered in LDS
// via async global->LDS b128 (one 16 B transfer per thread) so the DMA of
// tile k+1 overlaps the WMMA of tile k; bf16 conversion happens during
// fragment build.  B fragments come straight from pre-packed transposed
// bf16 weights (two uint4 loads per lane, L2 resident).
// ====================================================================
__global__ void __launch_bounds__(128)
spline_gemm_kernel(const float* __restrict__ acc,
                   const float* __restrict__ x,
                   const unsigned short* __restrict__ wT,
                   const unsigned short* __restrict__ rwT,
                   const float* __restrict__ bias,
                   const float* __restrict__ deg,
                   float* __restrict__ xconv) {
    __shared__ float sA[2][16 * 36];   // row stride 36 floats (16B-aligned)

    int tid  = threadIdx.x;
    int wave = tid >> 5;
    int lane = tid & 31;
    int m0   = blockIdx.x * 16;

    int hi   = (lane >= 16) ? 1 : 0;
    int row  = lane & 15;
    int colN = wave * 16 + row;

    // staging assignment: thread -> (row, 4-float column group)
    int srow = tid >> 3;
    int scol = (tid & 7) * 4;
    float* l0 = &sA[0][srow * 36 + scol];
    float* l1 = &sA[1][srow * 36 + scol];
    const float* gA = acc + (size_t)(m0 + srow) * KTOT + scol;

    v8f cc = {};   // conv accumulator
    v8f cr = {};   // root accumulator

    // -------- conv GEMM: acc[16 x 1728] x W[1728 x 64] --------
    stage_b128(gA, l0);                               // tile 0
    for (int kk = 0; kk < KSTEPS; ++kk) {
        int cur = kk & 1;
        if (kk + 1 < KSTEPS) {
            stage_b128(gA + (kk + 1) * 32, cur ? l0 : l1);   // next tile
            async_wait(1);                                    // tile kk landed
        } else {
            async_wait(0);
        }
        __syncthreads();

        v16bf a = load_a_frag(sA[cur], row, hi);
        v16bf b = load_b_frag(wT + (size_t)colN * KTOT + kk * 32 + hi * 16);
        cc = __builtin_amdgcn_wmma_f32_16x16x32_bf16(
                 false, a, false, b, (short)0, cc, false, false);
        __syncthreads();   // protect buffer before next-iteration staging
    }

    // -------- root GEMM: x[16 x 64] x root_w[64 x 64] --------
    const float* gX = x + (size_t)(m0 + srow) * CIN + scol;
    for (int kk = 0; kk < 2; ++kk) {
        stage_b128(gX + kk * 32, l0);
        async_wait(0);
        __syncthreads();

        v16bf a = load_a_frag(sA[0], row, hi);
        v16bf b = load_b_frag(rwT + (size_t)colN * CIN + kk * 32 + hi * 16);
        cr = __builtin_amdgcn_wmma_f32_16x16x32_bf16(
                 false, a, false, b, (short)0, cr, false, false);
        __syncthreads();
    }

    // -------- epilogue: deg divide + root + bias + ELU --------
    float bcol  = bias[colN];
    int   rbase = m0 + (hi ? 8 : 0);
#pragma unroll
    for (int v = 0; v < 8; ++v) {
        int node = rbase + v;
        float dg  = fmaxf(deg[node], 1.0f);
        float val = cc[v] / dg + cr[v] + bcol;
        val = (val > 0.0f) ? val : (__expf(val) - 1.0f);
        xconv[(size_t)node * COUT + colN] = val;
    }
}

// ====================================================================
// Kernel 4: radius neighbors. One wave32 per query; ascending-index scan
// with ballot+popcount compaction keeps exactly the first MAXN matches.
// ====================================================================
__global__ void __launch_bounds__(128)
radius_kernel(const float* __restrict__ pos,
              const int*   __restrict__ batch,
              const int*   __restrict__ idx,
              int* __restrict__ ncol,
              int* __restrict__ nrow) {
    int wglobal = blockIdx.x * (blockDim.x >> 5) + (threadIdx.x >> 5);
    int lane    = threadIdx.x & 31;
    if (wglobal >= MM) return;
    int q  = wglobal;
    int iq = idx[q];
    float qx = pos[iq * 3 + 0], qy = pos[iq * 3 + 1], qz = pos[iq * 3 + 2];
    int   bq = batch[iq];
    const float r2 = RRAD * RRAD;

    int count = 0;
    for (int j0 = 0; j0 < NN && count < MAXN; j0 += 32) {
        int j = j0 + lane;                         // NN is a multiple of 32
        float dx = pos[j * 3 + 0] - qx;
        float dy = pos[j * 3 + 1] - qy;
        float dz = pos[j * 3 + 2] - qz;
        float d2 = dx * dx + dy * dy + dz * dz;
        bool ok  = (d2 < r2) && (batch[j] == bq);
        unsigned m32 = (unsigned)__ballot(ok);
        int p = count + __popc(m32 & ((1u << lane) - 1u));
        if (ok && p < MAXN) {
            ncol[q * MAXN + p] = j;
            nrow[q * MAXN + p] = q;
        }
        count += __popc(m32);
    }
    count = min(count, MAXN);
    for (int s = count + lane; s < MAXN; s += 32) {
        ncol[q * MAXN + s] = -1;
        nrow[q * MAXN + s] = -1;
    }
}

// ====================================================================
// Kernel 5: gather selected rows into d_out.
// out layout (float slots): [0)=x_out M*64 | [320000)=col M*64 |
// [640000)=row M*64 | [960000)=pos M*3 | [975000)=batch M | [980000)=ea M*3
// ====================================================================
__global__ void gather_kernel(const float* __restrict__ xconv,
                              const float* __restrict__ pos,
                              const int*   __restrict__ batch,
                              const float* __restrict__ ea,
                              const int*   __restrict__ idx,
                              float* __restrict__ out) {
    int t = blockIdx.x * blockDim.x + threadIdx.x;
    if (t < MM * COUT) {
        int q = t >> 6, c = t & 63;
        out[t] = xconv[(size_t)idx[q] * COUT + c];
    }
    if (t < MM * 3) {
        int q = t / 3, d = t % 3;
        out[960000 + t] = pos[idx[q] * 3 + d];
        out[980000 + t] = ea[idx[q] * 3 + d];
    }
    if (t < MM) {
        ((int*)out)[975000 + t] = batch[idx[t]];
    }
}

// ====================================================================
extern "C" void kernel_launch(void* const* d_in, const int* in_sizes, int n_in,
                              void* d_out, int out_size, void* d_ws, size_t ws_size,
                              hipStream_t stream) {
    (void)in_sizes; (void)n_in; (void)out_size; (void)ws_size;
    const float* x     = (const float*)d_in[0];
    const int*   ei    = (const int*)  d_in[1];
    const float* ea    = (const float*)d_in[2];
    const float* pos   = (const float*)d_in[3];
    const int*   batch = (const int*)  d_in[4];
    const float* w     = (const float*)d_in[5];
    const float* rw    = (const float*)d_in[6];
    const float* bias  = (const float*)d_in[7];

    // workspace layout (all offsets stay 16B aligned)
    float* acc   = (float*)d_ws;                         // N*P*CIN
    float* deg   = acc + ACC_ELEMS;                      // N
    float* xconv = deg + NN;                             // N*COUT
    int*   idx   = (int*)(xconv + (size_t)NN * COUT);    // M
    unsigned short* wT  = (unsigned short*)(idx + MM);   // 64*1728 bf16
    unsigned short* rwT = wT + (size_t)KTOT * COUT;      // 64*64 bf16

    float* out = (float*)d_out;
    int* ncol = (int*)d_out + 320000;
    int* nrow = (int*)d_out + 640000;

    // 0) zero accumulator + degree, pack weights to transposed bf16
    zero_ws_kernel<<<4096, 256, 0, stream>>>((float4*)acc,
                                             (ACC_ELEMS + NN) / 4);
    pack_weights_kernel<<<((KTOT + CIN) * COUT + 255) / 256, 256, 0, stream>>>(
        w, rw, wT, rwT);

    // 1) edge scatter (one wave per edge)
    {
        long long threads = (long long)EE * 32;
        int blocks = (int)((threads + 255) / 256);
        spline_scatter_kernel<<<blocks, 256, 0, stream>>>(x, ei, ea, acc, deg);
    }

    // 2) FPS (independent of conv)
    fps_kernel<<<1, 1024, 0, stream>>>(pos, idx);

    // 3) WMMA GEMM: conv + root + bias + ELU
    spline_gemm_kernel<<<NN / 16, 128, 0, stream>>>(acc, x, wT, rwT, bias, deg,
                                                    xconv);

    // 4) radius neighbors (one wave per query)
    radius_kernel<<<(MM * 32 + 127) / 128, 128, 0, stream>>>(pos, batch, idx,
                                                             ncol, nrow);

    // 5) gather outputs
    gather_kernel<<<(MM * COUT + 255) / 256, 256, 0, stream>>>(xconv, pos, batch,
                                                               ea, idx, out);
}
